// GraphSAGERecommender_1039382086190
// MI455X (gfx1250) — matrix operand order repaired and checked by reading, not compiled, not run on hardware
//
#include <hip/hip_runtime.h>

#define N_NODES 100000
#define N_EDGES 1600000
#define CH 64
#define TILES (N_NODES / 16)   // 6250, exact

typedef __attribute__((ext_vector_type(16))) __bf16 v16bf;
typedef __attribute__((ext_vector_type(8)))  float  v8f;

// Native fp32 -> bf16 conversion: gfx1250 has first-class bf16 VALU support,
// so a plain cast lowers to v_cvt_*bf16* instead of a 3-op bit-trick.
__device__ __forceinline__ __bf16 f2bf(float f) { return (__bf16)f; }
__device__ __forceinline__ unsigned short f2bf_bits(float f) {
  __bf16 b = (__bf16)f;
  return __builtin_bit_cast(unsigned short, b);
}
__device__ __forceinline__ void atomic_add_f32(float* p, float v) {
  __hip_atomic_fetch_add(p, v, __ATOMIC_RELAXED, __HIP_MEMORY_SCOPE_AGENT);
}

// ---------------- utility kernels ----------------

__global__ void sage_zero_kernel(float* __restrict__ p, int n) {
  int i = blockIdx.x * blockDim.x + threadIdx.x;
  if (i < n) p[i] = 0.0f;
}

__global__ void sage_degree_kernel(const int* __restrict__ dst, float* __restrict__ deg) {
  int e = blockIdx.x * blockDim.x + threadIdx.x;
  if (e < N_EDGES) atomic_add_f32(&deg[dst[e]], 1.0f);
}

__global__ void sage_deginv_kernel(float* __restrict__ deg) {
  int i = blockIdx.x * blockDim.x + threadIdx.x;
  if (i < N_NODES) deg[i] = 1.0f / fmaxf(deg[i], 1.0f);
}

// Edge scatter: mean-aggregation numerator. 16 threads per edge, 4 channels each.
// Node table (25.6MB) is L2-resident on MI455X (192MB L2), so the random gather
// and the f32 atomic adds both stay in L2.
__global__ void sage_scatter_kernel(const float* __restrict__ h,
                                    const int* __restrict__ src,
                                    const int* __restrict__ dst,
                                    float* __restrict__ agg) {
  int t = blockIdx.x * blockDim.x + threadIdx.x;
  if (t >= N_EDGES * 16) return;
  int e  = t >> 4;
  int cg = (t & 15) << 2;
  int s = src[e];
  int d = dst[e];
  float4 v = *(const float4*)(h + (size_t)s * CH + cg);
  float* ap = agg + (size_t)d * CH + cg;
  atomic_add_f32(ap + 0, v.x);
  atomic_add_f32(ap + 1, v.y);
  atomic_add_f32(ap + 2, v.z);
  atomic_add_f32(ap + 3, v.w);
}

// ---------------- fused SAGE linear layer via WMMA ----------------
// out[m,n] = relu?( (agg[m,:] * dinv[m]) @ Wl^T + bl + h[m,:] @ Wr^T )
// One wave (32 lanes) per 16-row tile; 8 waves / 256-thread block.
// Both 64x64 weight matrices are pre-packed into LDS in exact
// V_WMMA_F32_16X16X32_BF16 B-fragment order (lane-major, 32B/lane/frag).
__global__ void __launch_bounds__(256)
sage_gemm_kernel(const float* __restrict__ agg,
                 const float* __restrict__ dinv,
                 const float* __restrict__ h,
                 const float* __restrict__ Wl,
                 const float* __restrict__ bl,
                 const float* __restrict__ Wr,
                 float* __restrict__ out,
                 int relu) {
  // 16 fragments: f = mat*8 + ntile*2 + kstep; each frag = 32 lanes * 16 halves
  __shared__ __align__(32) unsigned short sPack[16 * 512];

  // Cooperative pack: B[k][n] = W[n][k] in bf16, fragment-major layout.
  for (int idx = threadIdx.x; idx < 16 * 512; idx += 256) {
    int f    = idx >> 9;          // 0..15
    int r    = idx & 511;
    int lane = r >> 4;            // 0..31
    int i    = r & 15;            // half index within lane
    int mat   = f >> 3;           // 0 = Wl, 1 = Wr
    int ntile = (f >> 1) & 3;
    int kstep = f & 1;
    int n = ntile * 16 + (lane & 15);
    int k = kstep * 32 + ((lane >> 4) << 3) + (i & 7) + ((i >> 3) << 4);   // lanes 0-15: K 0..7,16..23, lanes 16-31: K 8..15,24..31
    const float* W = mat ? Wr : Wl;
    sPack[idx] = f2bf_bits(W[n * CH + k]);
  }
  __syncthreads();

  int wave = threadIdx.x >> 5;
  int lane = threadIdx.x & 31;
  int tile = blockIdx.x * 8 + wave;
  if (tile >= TILES) return;

  int row   = tile * 16 + (lane & 15);   // A-fragment row for this lane
  int kbase = (lane >> 4) << 3;          // 0 or 8 (16-bit A 16x32 layout)
  float di = dinv[row];

  v8f acc0 = {}, acc1 = {}, acc2 = {}, acc3 = {};

#pragma unroll
  for (int mat = 0; mat < 2; ++mat) {
    const float* A = mat ? h : agg;
    float scale = mat ? 1.0f : di;
    const float* rowp = A + (size_t)row * CH;
#pragma unroll
    for (int ks = 0; ks < 2; ++ks) {
      int k0 = ks * 32;
      // lane holds K = k0 + {kbase..kbase+7, 16+kbase..16+kbase+7}
      float4 r0 = *(const float4*)(rowp + k0 + kbase);
      float4 r1 = *(const float4*)(rowp + k0 + kbase + 4);
      float4 r2 = *(const float4*)(rowp + k0 + kbase + 16);
      float4 r3 = *(const float4*)(rowp + k0 + kbase + 20);
      v16bf a;
      a[0]  = f2bf(r0.x * scale); a[1]  = f2bf(r0.y * scale);
      a[2]  = f2bf(r0.z * scale); a[3]  = f2bf(r0.w * scale);
      a[4]  = f2bf(r1.x * scale); a[5]  = f2bf(r1.y * scale);
      a[6]  = f2bf(r1.z * scale); a[7]  = f2bf(r1.w * scale);
      a[8]  = f2bf(r2.x * scale); a[9]  = f2bf(r2.y * scale);
      a[10] = f2bf(r2.z * scale); a[11] = f2bf(r2.w * scale);
      a[12] = f2bf(r3.x * scale); a[13] = f2bf(r3.y * scale);
      a[14] = f2bf(r3.z * scale); a[15] = f2bf(r3.w * scale);

#pragma unroll
      for (int nt = 0; nt < 4; ++nt) {
        const v16bf b = *(const v16bf*)&sPack[((mat * 8 + nt * 2 + ks) << 9) + (lane << 4)];
        v8f c = (nt == 0) ? acc0 : (nt == 1) ? acc1 : (nt == 2) ? acc2 : acc3;
        c = __builtin_amdgcn_wmma_f32_16x16x32_bf16(
            /*neg_a=*/false, a, /*neg_b=*/false, b,
            /*c_mod=*/(short)0, c, /*reuse_a=*/false, /*reuse_b=*/false);
        if (nt == 0) acc0 = c; else if (nt == 1) acc1 = c;
        else if (nt == 2) acc2 = c; else acc3 = c;
      }
    }
  }

  // Epilogue: C/D layout — VGPR j: M = j (lanes 0-15) / j+8 (lanes 16-31); N = lane&15.
  int col   = lane & 15;
  int mrow0 = tile * 16 + ((lane >> 4) << 3);
#pragma unroll
  for (int nt = 0; nt < 4; ++nt) {
    int n = nt * 16 + col;
    float bias = bl[n];
    const v8f c = (nt == 0) ? acc0 : (nt == 1) ? acc1 : (nt == 2) ? acc2 : acc3;
#pragma unroll
    for (int j = 0; j < 8; ++j) {
      float v = c[j] + bias;
      if (relu) v = fmaxf(v, 0.0f);
      out[(size_t)(mrow0 + j) * CH + n] = v;
    }
  }
}

// ---------------- launch ----------------

extern "C" void kernel_launch(void* const* d_in, const int* in_sizes, int n_in,
                              void* d_out, int out_size, void* d_ws, size_t ws_size,
                              hipStream_t stream) {
  const float* x   = (const float*)d_in[0];
  const int*   ei  = (const int*)d_in[1];
  const float* Wl0 = (const float*)d_in[2];
  const float* bl0 = (const float*)d_in[3];
  const float* Wr0 = (const float*)d_in[4];
  const float* Wl1 = (const float*)d_in[5];
  const float* bl1 = (const float*)d_in[6];
  const float* Wr1 = (const float*)d_in[7];
  const float* Wl2 = (const float*)d_in[8];
  const float* bl2 = (const float*)d_in[9];
  const float* Wr2 = (const float*)d_in[10];
  float* out = (float*)d_out;

  const int* src = ei;
  const int* dst = ei + N_EDGES;

  float* dinv = (float*)d_ws;               // N floats (deg accumulated here first)
  float* agg  = dinv + N_NODES;             // N*64 floats
  float* h1   = agg + (size_t)N_NODES * CH; // N*64 floats

  const int ZB = (N_NODES * CH + 255) / 256;
  const int EB = (N_EDGES + 255) / 256;
  const int SB = (N_EDGES * 16 + 255) / 256;
  const int GB = (TILES + 7) / 8;

  // degrees -> deg_inv
  sage_zero_kernel<<<(N_NODES + 255) / 256, 256, 0, stream>>>(dinv, N_NODES);
  sage_degree_kernel<<<EB, 256, 0, stream>>>(dst, dinv);
  sage_deginv_kernel<<<(N_NODES + 255) / 256, 256, 0, stream>>>(dinv);

  // layer 0: x -> h1 (relu)
  sage_zero_kernel<<<ZB, 256, 0, stream>>>(agg, N_NODES * CH);
  sage_scatter_kernel<<<SB, 256, 0, stream>>>(x, src, dst, agg);
  sage_gemm_kernel<<<GB, 256, 0, stream>>>(agg, dinv, x, Wl0, bl0, Wr0, h1, 1);

  // layer 1: h1 -> out (relu)
  sage_zero_kernel<<<ZB, 256, 0, stream>>>(agg, N_NODES * CH);
  sage_scatter_kernel<<<SB, 256, 0, stream>>>(h1, src, dst, agg);
  sage_gemm_kernel<<<GB, 256, 0, stream>>>(agg, dinv, h1, Wl1, bl1, Wr1, out, 1);

  // layer 2: out -> out in place (no relu); each wave reads only its own rows
  sage_zero_kernel<<<ZB, 256, 0, stream>>>(agg, N_NODES * CH);
  sage_scatter_kernel<<<SB, 256, 0, stream>>>(out, src, dst, agg);
  sage_gemm_kernel<<<GB, 256, 0, stream>>>(agg, dinv, out, Wl2, bl2, Wr2, out, 0);
}
